// Processor_343597383944
// MI455X (gfx1250) — compile-verified, matrix operand order checked by reading, not run on hardware
//
#include <hip/hip_runtime.h>
#include <math.h>

typedef __attribute__((ext_vector_type(16))) _Float16 v16h;
typedef __attribute__((ext_vector_type(8)))  _Float16 v8h;
typedef __attribute__((ext_vector_type(4)))  _Float16 v4h;
typedef __attribute__((ext_vector_type(8)))  float    v8f;

#define D128 128

// ---------- helpers ----------
__device__ __forceinline__ unsigned int f2ord(float f) {
    unsigned int u = __float_as_uint(f);
    return (u & 0x80000000u) ? ~u : (u | 0x80000000u);
}
__device__ __forceinline__ float ord2f(unsigned int e) {
    unsigned int u = (e & 0x80000000u) ? (e ^ 0x80000000u) : ~e;
    return __uint_as_float(u);
}
__device__ __forceinline__ float wave_sum(float v) {
    #pragma unroll
    for (int off = 16; off > 0; off >>= 1) v += __shfl_xor(v, off, 32);
    return v;
}

// ---------- one-time: x (f32) -> x16 (f16) ----------
__global__ void conv_f16(const float* __restrict__ src, _Float16* __restrict__ dst, int n4) {
    const int i = blockIdx.x * blockDim.x + threadIdx.x;
    if (i >= n4) return;
    const float4 v = ((const float4*)src)[i];
    v4h h; h[0] = (_Float16)v.x; h[1] = (_Float16)v.y; h[2] = (_Float16)v.z; h[3] = (_Float16)v.w;
    *(v4h*)(dst + 4 * (size_t)i) = h;
}

// ---------- one-time: Wt16[l][n][k] = (f16) W[l][k][n] ----------
__global__ void prep_wt(const float* __restrict__ W, _Float16* __restrict__ Wt, int total) {
    const int t = blockIdx.x * blockDim.x + threadIdx.x;     // total = 3*128*32
    if (t >= total) return;
    const int k4 = (t & 31) * 4;
    const int n  = (t >> 5) & 127;
    const int l  = t >> 12;
    const float* wp = W + (size_t)l * D128 * D128 + (size_t)k4 * D128 + n;
    v4h h;
    h[0] = (_Float16)wp[0 * D128];
    h[1] = (_Float16)wp[1 * D128];
    h[2] = (_Float16)wp[2 * D128];
    h[3] = (_Float16)wp[3 * D128];
    *(v4h*)(Wt + ((size_t)l * D128 + n) * D128 + k4) = h;
}

// ---------- GEMM: C[n,128] = A16[n,128] @ Wt16^T, WMMA f16->f32 ----------
__global__ __launch_bounds__(256) void gemm128(const _Float16* __restrict__ A16,
                                               const _Float16* __restrict__ Wt16,
                                               float* __restrict__ C, int nrows) {
    const int wave = threadIdx.x >> 5;
    const int lane = threadIdx.x & 31;
    const int half = lane >> 4;       // 0: lanes 0-15, 1: lanes 16-31
    const int l15  = lane & 15;
    const int rowBase = blockIdx.x * 16;
    const int colBase = wave * 16;
    const int kOff = half * 8;

    int row = rowBase + l15; if (row >= nrows) row = nrows - 1;
    const int col = colBase + l15;

    const _Float16* arow = A16 + (size_t)row * D128;
    const _Float16* brow = Wt16 + (size_t)col * D128;

    v8f acc = {};
    #pragma unroll
    for (int k0 = 0; k0 < 128; k0 += 32) {
        const v8h alo = *(const v8h*)(arow + k0 + kOff);
        const v8h ahi = *(const v8h*)(arow + k0 + 16 + kOff);
        const v8h blo = *(const v8h*)(brow + k0 + kOff);
        const v8h bhi = *(const v8h*)(brow + k0 + 16 + kOff);
        v16h a, b;
        #pragma unroll
        for (int i = 0; i < 8; ++i) {
            a[i] = alo[i]; a[8 + i] = ahi[i];
            b[i] = blo[i]; b[8 + i] = bhi[i];
        }
        acc = __builtin_amdgcn_wmma_f32_16x16x32_f16(false, a, false, b,
                                                     (short)0, acc, false, false);
    }
    const int mOff = half * 8;
    float* crow = C + (size_t)(rowBase + mOff) * D128 + colBase + l15;
    if (rowBase + 16 <= nrows) {
        // fast path: straight-line stores, immediate offsets
        #pragma unroll
        for (int r = 0; r < 8; ++r) crow[(size_t)r * D128] = acc[r];
    } else {
        #pragma unroll
        for (int r = 0; r < 8; ++r)
            if (rowBase + mOff + r < nrows) crow[(size_t)r * D128] = acc[r];
    }
}

// ---------- we_vec[l][j] = sum_d W_e[l][j][d] * att_edge[l][d] ----------
__global__ __launch_bounds__(256) void wevec_kernel(const float* __restrict__ We,
                                                    const float* __restrict__ ae,
                                                    float* __restrict__ wevec, int nrows) {
    const int wv = blockIdx.x * (blockDim.x >> 5) + (threadIdx.x >> 5);
    if (wv >= nrows) return;              // nrows = 3*128
    const int lane = threadIdx.x & 31;
    const int l = wv >> 7, j = wv & 127;
    const float* wr = We + ((size_t)l * D128 + j) * D128 + lane * 4;
    const float* ar = ae + (size_t)l * D128 + lane * 4;
    float s = wr[0]*ar[0] + wr[1]*ar[1] + wr[2]*ar[2] + wr[3]*ar[3];
    s = wave_sum(s);
    if (lane == 0) wevec[wv] = s;
}

// ---------- per-edge scalars for all 3 layers + deg / per-dst sums ----------
__global__ __launch_bounds__(256) void edge_scalar(const float* __restrict__ ea,
                                                   const float* __restrict__ wevec,
                                                   const int* __restrict__ dst,
                                                   float* __restrict__ sedge,
                                                   float* __restrict__ sums,
                                                   float* __restrict__ deg,
                                                   int E, int N) {
    const int e = blockIdx.x * (blockDim.x >> 5) + (threadIdx.x >> 5);
    if (e >= E) return;
    const int lane = threadIdx.x & 31;
    const float* er = ea + (size_t)e * D128 + lane * 4;
    const float e0 = er[0], e1 = er[1], e2 = er[2], e3 = er[3];
    float s[3];
    #pragma unroll
    for (int l = 0; l < 3; ++l) {
        const float* wv = wevec + l * D128 + lane * 4;
        s[l] = wave_sum(e0*wv[0] + e1*wv[1] + e2*wv[2] + e3*wv[3]);
    }
    if (lane == 0) {
        const int d = dst[e];
        #pragma unroll
        for (int l = 0; l < 3; ++l) {
            sedge[(size_t)l * E + e] = s[l];
            atomicAdd(&sums[(size_t)l * N + d], s[l]);
        }
        atomicAdd(&deg[d], 1.0f);
    }
}

__global__ void loop_kernel(const float* __restrict__ sums, const float* __restrict__ deg,
                            float* __restrict__ loops, int N) {
    const int i = blockIdx.x * blockDim.x + threadIdx.x;
    if (i >= N) return;
    const float dinv = 1.0f / fmaxf(deg[i], 1.0f);
    #pragma unroll
    for (int l = 0; l < 3; ++l) loops[(size_t)l * N + i] = sums[(size_t)l * N + i] * dinv;
}

// ---------- per-node attention dots: as[i]=h[i].att_src, ad[i]=h[i].att_dst ----------
__global__ __launch_bounds__(256) void node_dots(const float* __restrict__ h,
                                                 const float* __restrict__ atts,
                                                 const float* __restrict__ attd,
                                                 float* __restrict__ as_, float* __restrict__ ad_,
                                                 int N) {
    const int i = blockIdx.x * (blockDim.x >> 5) + (threadIdx.x >> 5);
    if (i >= N) return;
    const int lane = threadIdx.x & 31;
    const float* hr = h + (size_t)i * D128 + lane * 4;
    const float* sr = atts + lane * 4;
    const float* dr = attd + lane * 4;
    float s = hr[0]*sr[0] + hr[1]*sr[1] + hr[2]*sr[2] + hr[3]*sr[3];
    float d = hr[0]*dr[0] + hr[1]*dr[1] + hr[2]*dr[2] + hr[3]*dr[3];
    s = wave_sum(s); d = wave_sum(d);
    if (lane == 0) { as_[i] = s; ad_[i] = d; }
}

__global__ void fill_u32(unsigned int* __restrict__ p, unsigned int v, int n) {
    const int i = blockIdx.x * blockDim.x + threadIdx.x;
    if (i < n) p[i] = v;
}

// ---------- pass A: logits + LeakyReLU + segment max (ordered-uint atomicMax) ----------
__global__ void passA(const int* __restrict__ src, const int* __restrict__ dst,
                      const float* __restrict__ as_, const float* __restrict__ ad_,
                      const float* __restrict__ sedge_l, const float* __restrict__ loops_l,
                      float* __restrict__ abuf, unsigned int* __restrict__ menc,
                      int E, int NE) {
    const int idx = blockIdx.x * blockDim.x + threadIdx.x;
    if (idx >= NE) return;
    int s, d; float sv;
    if (idx < E) { s = src[idx]; d = dst[idx]; sv = sedge_l[idx]; }
    else         { s = d = idx - E;            sv = loops_l[s];  }
    float a = as_[s] + ad_[d] + sv;
    a = (a > 0.0f) ? a : 0.2f * a;
    abuf[idx] = a;
    atomicMax(&menc[d], f2ord(a));
}

// ---------- pass B: ea = exp(a - m[dst]); denom[dst] += ea ----------
__global__ void passB(const int* __restrict__ dst,
                      float* __restrict__ abuf, const unsigned int* __restrict__ menc,
                      float* __restrict__ denom, int E, int NE) {
    const int idx = blockIdx.x * blockDim.x + threadIdx.x;
    if (idx >= NE) return;
    const int d = (idx < E) ? dst[idx] : (idx - E);
    const float ea = __expf(abuf[idx] - ord2f(menc[d]));
    abuf[idx] = ea;
    atomicAdd(&denom[d], ea);
}

// ---------- pass C: agg[dst] += (ea/denom[dst]) * h[src] ----------
__global__ __launch_bounds__(256) void passC(const int* __restrict__ src, const int* __restrict__ dst,
                                             const float* __restrict__ abuf,
                                             const float* __restrict__ denom,
                                             const float* __restrict__ h,
                                             float* __restrict__ agg, int E, int NE) {
    const int idx = blockIdx.x * (blockDim.x >> 5) + (threadIdx.x >> 5);
    if (idx >= NE) return;
    const int lane = threadIdx.x & 31;
    int s, d;
    if (idx < E) { s = src[idx]; d = dst[idx]; } else { s = d = idx - E; }
    const float alpha = abuf[idx] / denom[d];
    const float* hr = h + (size_t)s * D128 + lane * 4;
    float* ar = agg + (size_t)d * D128 + lane * 4;
    atomicAdd(&ar[0], alpha * hr[0]);
    atomicAdd(&ar[1], alpha * hr[1]);
    atomicAdd(&ar[2], alpha * hr[2]);
    atomicAdd(&ar[3], alpha * hr[3]);
}

// ---------- bias + LayerNorm; writes f16 (mid layers) or f32 (final) ----------
__global__ __launch_bounds__(256) void ln_kernel(const float* __restrict__ agg,
                                                 const float* __restrict__ bias,
                                                 const float* __restrict__ gamma,
                                                 const float* __restrict__ beta,
                                                 float* __restrict__ outf,
                                                 _Float16* __restrict__ outh,
                                                 int N) {
    const int i = blockIdx.x * (blockDim.x >> 5) + (threadIdx.x >> 5);
    if (i >= N) return;
    const int lane = threadIdx.x & 31;
    const float* ar = agg + (size_t)i * D128 + lane * 4;
    const float* br = bias + lane * 4;
    float v0 = ar[0] + br[0], v1 = ar[1] + br[1], v2 = ar[2] + br[2], v3 = ar[3] + br[3];
    const float mu = wave_sum(v0 + v1 + v2 + v3) * (1.0f / 128.0f);
    v0 -= mu; v1 -= mu; v2 -= mu; v3 -= mu;
    const float var = wave_sum(v0*v0 + v1*v1 + v2*v2 + v3*v3) * (1.0f / 128.0f);
    const float rs = rsqrtf(var + 1e-5f);
    const float* gr = gamma + lane * 4;
    const float* er = beta + lane * 4;
    const float o0 = v0 * rs * gr[0] + er[0];
    const float o1 = v1 * rs * gr[1] + er[1];
    const float o2 = v2 * rs * gr[2] + er[2];
    const float o3 = v3 * rs * gr[3] + er[3];
    if (outh) {
        v4h h; h[0] = (_Float16)o0; h[1] = (_Float16)o1; h[2] = (_Float16)o2; h[3] = (_Float16)o3;
        *(v4h*)(outh + (size_t)i * D128 + lane * 4) = h;
    } else {
        float* orow = outf + (size_t)i * D128 + lane * 4;
        orow[0] = o0; orow[1] = o1; orow[2] = o2; orow[3] = o3;
    }
}

// ---------- host ----------
static inline size_t alignup(size_t x) { return (x + 255) & ~(size_t)255; }

extern "C" void kernel_launch(void* const* d_in, const int* in_sizes, int n_in,
                              void* d_out, int out_size, void* d_ws, size_t ws_size,
                              hipStream_t stream) {
    const float* x        = (const float*)d_in[0];
    const int*   ei       = (const int*)  d_in[1];
    const float* edgeattr = (const float*)d_in[2];
    const float* W        = (const float*)d_in[3];
    const float* att_src  = (const float*)d_in[4];
    const float* att_dst  = (const float*)d_in[5];
    const float* W_e      = (const float*)d_in[6];
    const float* att_edge = (const float*)d_in[7];
    const float* bias     = (const float*)d_in[8];
    const float* gamma    = (const float*)d_in[9];
    const float* beta     = (const float*)d_in[10];

    const int N  = in_sizes[0] / D128;
    const int E  = in_sizes[1] / 2;
    const int NE = E + N;
    const int* src = ei;
    const int* dst = ei + E;

    // carve workspace
    char* p = (char*)d_ws;
    auto take = [&](size_t bytes) { char* r = p; p += alignup(bytes); return r; };
    float*     hlin  = (float*)    take((size_t)N * D128 * 4);
    float*     agg   = (float*)    take((size_t)N * D128 * 4);
    _Float16*  x16   = (_Float16*) take((size_t)N * D128 * 2);
    _Float16*  hio16 = (_Float16*) take((size_t)N * D128 * 2);
    _Float16*  wt16  = (_Float16*) take((size_t)3 * D128 * D128 * 2);
    float*     asb   = (float*)    take((size_t)N * 4);
    float*     adb   = (float*)    take((size_t)N * 4);
    float*     wevec = (float*)    take((size_t)3 * D128 * 4);
    float*     sedge = (float*)    take((size_t)3 * E * 4);
    float*     deg   = (float*)    take((size_t)N * 4);
    float*     sums  = (float*)    take((size_t)3 * N * 4);
    float*     loops = (float*)    take((size_t)3 * N * 4);
    float*     abuf  = (float*)    take((size_t)NE * 4);
    unsigned int* menc = (unsigned int*)take((size_t)N * 4);
    float*     denom = (float*)    take((size_t)N * 4);
    (void)ws_size; (void)n_in; (void)out_size;

    const int TB = 256;
    const int WPB = TB / 32;

    // one-time precompute
    hipMemsetAsync(deg,  0, (size_t)N * 4, stream);
    hipMemsetAsync(sums, 0, (size_t)3 * N * 4, stream);
    conv_f16<<<((N * D128 / 4) + TB - 1) / TB, TB, 0, stream>>>(x, x16, N * D128 / 4);
    prep_wt<<<(3 * D128 * 32 + TB - 1) / TB, TB, 0, stream>>>(W, wt16, 3 * D128 * 32);
    wevec_kernel<<<(3 * D128 + WPB - 1) / WPB, TB, 0, stream>>>(W_e, att_edge, wevec, 3 * D128);
    edge_scalar<<<(E + WPB - 1) / WPB, TB, 0, stream>>>(edgeattr, wevec, dst, sedge, sums, deg, E, N);
    loop_kernel<<<(N + TB - 1) / TB, TB, 0, stream>>>(sums, deg, loops, N);

    const unsigned int ORD_NEG_INF = 0x007FFFFFu; // f2ord(-inf)

    const _Float16* hin16 = x16;
    for (int l = 0; l < 3; ++l) {
        gemm128<<<(N + 15) / 16, TB, 0, stream>>>(hin16, wt16 + (size_t)l * D128 * D128, hlin, N);
        node_dots<<<(N + WPB - 1) / WPB, TB, 0, stream>>>(hlin, att_src + l * D128,
                                                          att_dst + l * D128, asb, adb, N);
        fill_u32<<<(N + TB - 1) / TB, TB, 0, stream>>>(menc, ORD_NEG_INF, N);
        hipMemsetAsync(denom, 0, (size_t)N * 4, stream);
        hipMemsetAsync(agg,   0, (size_t)N * D128 * 4, stream);

        passA<<<(NE + TB - 1) / TB, TB, 0, stream>>>(src, dst, asb, adb,
                                                     sedge + (size_t)l * E, loops + (size_t)l * N,
                                                     abuf, menc, E, NE);
        passB<<<(NE + TB - 1) / TB, TB, 0, stream>>>(dst, abuf, menc, denom, E, NE);
        passC<<<(NE + WPB - 1) / WPB, TB, 0, stream>>>(src, dst, abuf, denom, hlin, agg, E, NE);

        if (l == 2) {
            ln_kernel<<<(N + WPB - 1) / WPB, TB, 0, stream>>>(agg, bias + l * D128,
                                                              gamma + l * D128, beta + l * D128,
                                                              (float*)d_out, nullptr, N);
        } else {
            ln_kernel<<<(N + WPB - 1) / WPB, TB, 0, stream>>>(agg, bias + l * D128,
                                                              gamma + l * D128, beta + l * D128,
                                                              nullptr, hio16, N);
        }
        hin16 = hio16;
    }
}
